// DiT_18322330485452
// MI455X (gfx1250) — matrix operand order repaired and checked by reading, not compiled, not run on hardware
//
#include <hip/hip_runtime.h>
#include <math.h>

typedef _Float16 half_t;
typedef __attribute__((ext_vector_type(16))) _Float16 v16h;
typedef __attribute__((ext_vector_type(8)))  float    v8f;

#define GENL   512
#define HISTL  512
#define NSEQ   1024
#define CDIM   1152
#define LATD   32
#define NHEAD  16
#define DHEAD  72
#define DHP    96
#define NB     4
#define C6     (6*CDIM)
#define NROWS  (NB*NSEQ)      // 4096
#define MCOND  (HISTL+NB)     // 516 distinct conditioning rows

__device__ __forceinline__ float siluf(float v){ return v/(1.f+__expf(-v)); }
__device__ __forceinline__ float geluf(float v){
    float u = 0.7978845608028654f*(v+0.044715f*v*v*v);
    return 0.5f*v*(1.f+tanhf(u));
}
// conditioning row map: HIST rows are batch-independent, GEN rows depend only on b
__device__ __forceinline__ int mrow_of(int gr){
    int n = gr & (NSEQ-1);
    return (n < HISTL) ? n : (HISTL + (gr >> 10));
}

// ---------------------------------------------------------------------------
// WMMA GEMM: out = epilogue( scale * A[f16 MxK] @ B[f16 KxN] )
// 256 threads = 8 waves; block tile 128x128, K-step 32; wave tile 32x64.
// A tile is staged with CDNA5 async global->LDS copies (ASYNCcnt) when the
// tile is fully in range; boundary tiles fall back to guarded register copies.
// Epilogue: +bias[col], act (0 none / 1 silu / 2 gelu), *gate, +res,
//           store f32 and/or f16.  Batched over blockIdx.z via sA/sB/sF.
// ---------------------------------------------------------------------------
#define BM 128
#define BN 128
#define KT 32
#define LDS_A 40
#define LDS_B 40

__global__ __launch_bounds__(256)
void gemm_wmma(const half_t* __restrict__ A, long long sA, int lda,
               const half_t* __restrict__ B, long long sB, int ldb,
               const float* __restrict__ bias,
               const float* __restrict__ gate, int ldg, int gmap,
               const float* __restrict__ res, int ldr,
               float* __restrict__ oF, long long sF, int ldoF,
               half_t* __restrict__ oH, int ldoH,
               int M, int N, int K, int act, float scale)
{
    __shared__ half_t As[BM*LDS_A];
    __shared__ half_t Bs[BN*LDS_B];

    A += (long long)blockIdx.z * sA;
    B += (long long)blockIdx.z * sB;
    const long long zF = (long long)blockIdx.z * sF;

    const int m0 = blockIdx.y * BM;
    const int n0 = blockIdx.x * BN;
    const int tid  = threadIdx.x;
    const int lane = tid & 31;
    const int wave = tid >> 5;
    const int wm = wave >> 1;        // 0..3 -> M offset wm*32
    const int wn = wave & 1;         // 0..1 -> N offset wn*64
    const int rlane = lane & 15;
    const int kh    = lane >> 4;     // lane half

    const bool fullM = (m0 + BM) <= M;

    // per-thread A-tile chunk coordinates (2 chunks of 8 halves each)
    const int arow0 = tid >> 2,         acol0 = (tid & 3) * 8;
    const int arow1 = (tid + 256) >> 2, acol1 = (tid & 3) * 8;

    v8f acc[2][4] = {};

    for (int kt = 0; kt < K; kt += KT) {
        // --- A tile: 128x32 halves ---
        if (fullM) {
            // CDNA5 async DMA: global -> LDS, no VGPR staging (ASYNCcnt)
            const half_t* g0 = A + (long long)(m0 + arow0)*lda + kt + acol0;
            const half_t* g1 = A + (long long)(m0 + arow1)*lda + kt + acol1;
            unsigned l0 = (unsigned)(uintptr_t)(&As[arow0*LDS_A + acol0]);
            unsigned l1 = (unsigned)(uintptr_t)(&As[arow1*LDS_A + acol1]);
            asm volatile("global_load_async_to_lds_b128 %0, %1, off"
                         :: "v"(l0), "v"(g0) : "memory");
            asm volatile("global_load_async_to_lds_b128 %0, %1, off"
                         :: "v"(l1), "v"(g1) : "memory");
        } else {
            #pragma unroll
            for (int it = 0; it < 2; ++it) {
                int row  = it ? arow1 : arow0;
                int col8 = it ? acol1 : acol0;
                int gr   = m0 + row;
                union { uint4 u; half_t h[8]; } ub;
                ub.u = uint4{0u,0u,0u,0u};
                if (gr < M) ub.u = *(const uint4*)(A + (long long)gr*lda + kt + col8);
                *(uint4*)(&As[row*LDS_A + col8]) = ub.u;
            }
        }
        // --- B tile: 32x128, stored transposed Bs[n][k] ---
        #pragma unroll
        for (int it = 0; it < 2; ++it) {
            int c = tid + it*256;
            int krow  = c >> 4;
            int ncol8 = (c & 15) * 8;
            int gn    = n0 + ncol8;
            union { uint4 u; half_t h[8]; } ub;
            ub.u = uint4{0u,0u,0u,0u};
            if (gn < N) ub.u = *(const uint4*)(B + (long long)(kt+krow)*ldb + gn);
            #pragma unroll
            for (int j = 0; j < 8; ++j)
                Bs[(ncol8+j)*LDS_B + krow] = ub.h[j];
        }
        if (kt + KT < K) {   // speculative prefetch of next K tile
            __builtin_prefetch(A + (long long)(m0 + (tid >> 1))*lda + kt + KT, 0, 0);
            __builtin_prefetch(B + (long long)(kt + KT + (tid & 31))*ldb + n0, 0, 0);
        }
        // each wave drains its own async LDS writes before signalling the barrier
        asm volatile("s_wait_asynccnt 0x0" ::: "memory");
        __syncthreads();

        // --- fragments (contiguous k = kh*16 + e for both A and B)
        v16h af[2], bf[4];
        #pragma unroll
        for (int mi = 0; mi < 2; ++mi) {
            const half_t* src = &As[(wm*32 + mi*16 + rlane)*LDS_A];
            #pragma unroll
            for (int e = 0; e < 16; ++e) {
                int k = kh*16 + e;
                af[mi][e] = src[k];
            }
        }
        #pragma unroll
        for (int ni = 0; ni < 4; ++ni) {
            const half_t* src = &Bs[(wn*64 + ni*16 + rlane)*LDS_B + kh*16];
            #pragma unroll
            for (int e = 0; e < 16; ++e) bf[ni][e] = src[e];
        }
        #pragma unroll
        for (int mi = 0; mi < 2; ++mi)
            #pragma unroll
            for (int ni = 0; ni < 4; ++ni)
                acc[mi][ni] = __builtin_amdgcn_wmma_f32_16x16x32_f16(
                    false, af[mi], false, bf[ni], (short)0, acc[mi][ni], false, false);
        __syncthreads();
    }

    // --- epilogue
    #pragma unroll
    for (int mi = 0; mi < 2; ++mi) {
        #pragma unroll
        for (int ni = 0; ni < 4; ++ni) {
            int bc = n0 + wn*64 + ni*16 + rlane;
            if (bc >= N) continue;
            int base = m0 + wm*32 + mi*16 + kh*8;
            #pragma unroll
            for (int r = 0; r < 8; ++r) {
                int gr = base + r;
                if (gr >= M) continue;
                float v = acc[mi][ni][r] * scale;
                if (bias) v += bias[bc];
                if (act == 1) v = siluf(v);
                else if (act == 2) v = geluf(v);
                if (gate) {
                    int grow = gmap ? mrow_of(gr) : gr;
                    v *= gate[(long long)grow*ldg + bc];
                }
                if (res) v += res[(long long)gr*ldr + bc];
                if (oF)  oF[zF + (long long)gr*ldoF + bc] = v;
                if (oH)  oH[(long long)gr*ldoH + bc] = (half_t)v;
            }
        }
    }
}

// ---------------------------------------------------------------------------
// Elementwise / reduction kernels
// ---------------------------------------------------------------------------
__global__ void k_convert(const float* __restrict__ s, half_t* __restrict__ d, long long n) {
    long long i = (long long)blockIdx.x*256 + threadIdx.x;
    if (i < n) d[i] = (half_t)s[i];
}

__global__ void k_emb(const float* __restrict__ noise, half_t* __restrict__ emb) {
    int b = blockIdx.x, j = threadIdx.x;           // j in [0,128)
    float t = noise[b];
    float f = __expf(-9.210340371976184f * (float)j / 128.f);  // 1/10000^(j/128)
    float a = t * f;
    emb[b*256 + j]       = (half_t)cosf(a);
    emb[b*256 + 128 + j] = (half_t)sinf(a);
}

__global__ void k_silu_plug(const float* __restrict__ plug, half_t* __restrict__ dst) {
    long long i = (long long)blockIdx.x*256 + threadIdx.x;
    if (i < (long long)HISTL*CDIM) dst[i] = (half_t)siluf(plug[i]);
}

__global__ void k_xinit(const float* __restrict__ xin, const float* __restrict__ rawh,
                        const float* __restrict__ pos, float* __restrict__ x) {
    long long i = (long long)blockIdx.x*256 + threadIdx.x;
    if (i >= (long long)NROWS*CDIM) return;
    int gr = (int)(i / CDIM), c = (int)(i % CDIM);
    int b = gr >> 10, n = gr & (NSEQ-1);
    int lt = c % LATD;
    float v = (n < HISTL) ? rawh[((long long)(b*HISTL + n))*LATD + lt]
                          : xin [((long long)(b*GENL + (n-HISTL)))*LATD + lt];
    x[i] = v + pos[(long long)n*CDIM + c];
}

__global__ __launch_bounds__(256)
void k_lnmod(const float* __restrict__ x, const float* __restrict__ mod,
             half_t* __restrict__ y, int shoff) {
    int gr = blockIdx.x, tid = threadIdx.x;
    const float* xr = x + (long long)gr*CDIM;
    float s = 0.f, ss = 0.f;
    for (int c = tid; c < CDIM; c += 256) { float v = xr[c]; s += v; ss += v*v; }
    __shared__ float r1[256], r2[256];
    r1[tid] = s; r2[tid] = ss; __syncthreads();
    for (int o = 128; o > 0; o >>= 1) {
        if (tid < o) { r1[tid] += r1[tid+o]; r2[tid] += r2[tid+o]; }
        __syncthreads();
    }
    float mean = r1[0] * (1.f/CDIM);
    float var  = r2[0] * (1.f/CDIM) - mean*mean;
    float rs   = rsqrtf(var + 1e-6f);
    const float* mo = mod + (long long)mrow_of(gr)*C6 + shoff;
    half_t* yr = y + (long long)gr*CDIM;
    for (int c = tid; c < CDIM; c += 256) {
        float v = (xr[c] - mean) * rs;
        yr[c] = (half_t)(v * (1.f + mo[CDIM + c]) + mo[c]);
    }
}

__global__ void k_repack_qkv(const float* __restrict__ qkv, half_t* __restrict__ qp,
                             half_t* __restrict__ kt, half_t* __restrict__ vp) {
    long long i = (long long)blockIdx.x*256 + threadIdx.x;
    if (i >= (long long)NB*NHEAD*NSEQ*DHP) return;
    int d = (int)(i % DHP);
    int n = (int)((i / DHP) & (NSEQ-1));
    int h = (int)((i / ((long long)DHP*NSEQ)) & (NHEAD-1));
    int b = (int)(i / ((long long)DHP*NSEQ*NHEAD));
    const float* row = qkv + ((long long)(b*NSEQ + n)) * (3*CDIM);
    half_t qv = (half_t)0.f, kv = (half_t)0.f, vv = (half_t)0.f;
    if (d < DHEAD) {
        qv = (half_t)row[           h*DHEAD + d];
        kv = (half_t)row[  CDIM   + h*DHEAD + d];
        vv = (half_t)row[2*CDIM   + h*DHEAD + d];
    }
    long long hb = (long long)(b*NHEAD + h);
    qp[(hb*NSEQ + n)*DHP + d] = qv;
    vp[(hb*NSEQ + n)*DHP + d] = vv;
    kt[(hb*DHP + d)*NSEQ + n] = kv;      // transposed (d-major) for QK^T B operand
}

__global__ __launch_bounds__(256)
void k_softmax(float* __restrict__ S, const int* __restrict__ mask) {
    int row = blockIdx.x, tid = threadIdx.x;
    int i = row & (NSEQ-1);
    float* Sr = S + (long long)row*NSEQ;
    const int* mr = mask + (long long)i*NSEQ;
    float vals[4]; int mk[4];
    float mx = -1e30f;
    #pragma unroll
    for (int p = 0; p < 4; ++p) {
        int j = tid + p*256;
        mk[p] = mr[j];
        float v = mk[p] ? Sr[j] : -1e30f;
        vals[p] = v; mx = fmaxf(mx, v);
    }
    __shared__ float red[256];
    red[tid] = mx; __syncthreads();
    for (int o = 128; o > 0; o >>= 1) { if (tid < o) red[tid] = fmaxf(red[tid], red[tid+o]); __syncthreads(); }
    mx = red[0]; __syncthreads();
    float s = 0.f;
    #pragma unroll
    for (int p = 0; p < 4; ++p) {
        float e = mk[p] ? __expf(vals[p]-mx) : 0.f;
        vals[p] = e; s += e;
    }
    red[tid] = s; __syncthreads();
    for (int o = 128; o > 0; o >>= 1) { if (tid < o) red[tid] += red[tid+o]; __syncthreads(); }
    float inv = 1.f / red[0];
    half_t* Ph = (half_t*)Sr;        // in-place f32 -> f16 (row-local, regs cached)
    #pragma unroll
    for (int p = 0; p < 4; ++p) Ph[tid + p*256] = (half_t)(vals[p]*inv);
}

__global__ void k_repack_o(const float* __restrict__ op, half_t* __restrict__ oh) {
    long long i = (long long)blockIdx.x*256 + threadIdx.x;
    if (i >= (long long)NROWS*CDIM) return;
    int gr = (int)(i / CDIM), c = (int)(i % CDIM);
    int b = gr >> 10, n = gr & (NSEQ-1);
    int h = c / DHEAD, d = c % DHEAD;
    oh[i] = (half_t)op[(((long long)(b*NHEAD + h))*NSEQ + n)*DHP + d];
}

__global__ void k_finmodulate(const float* __restrict__ x, const float* __restrict__ fm,
                              half_t* __restrict__ xs) {
    long long i = (long long)blockIdx.x*256 + threadIdx.x;
    if (i >= (long long)NROWS*CDIM) return;
    int gr = (int)(i / CDIM), c = (int)(i % CDIM);
    const float* m = fm + (long long)mrow_of(gr)*(2*CDIM);
    xs[i] = (half_t)(x[i] * (1.f + m[CDIM + c]) + m[c]);
}

__global__ void k_slice(const float* __restrict__ of, float* __restrict__ out) {
    long long i = (long long)blockIdx.x*256 + threadIdx.x;
    if (i >= (long long)NB*GENL*LATD) return;
    int l = (int)(i % LATD);
    int g = (int)((i / LATD) % GENL);
    int b = (int)(i / ((long long)LATD*GENL));
    out[i] = of[((long long)(b*NSEQ + HISTL + g))*LATD + l];
}

// ---------------------------------------------------------------------------
// Host side
// ---------------------------------------------------------------------------
static inline void launch_gemm(hipStream_t st,
    const half_t* A, long long sA, int lda,
    const half_t* B, long long sB, int ldb,
    const float* bias,
    const float* gate, int ldg, int gmap,
    const float* res, int ldr,
    float* oF, long long sF, int ldoF,
    half_t* oH, int ldoH,
    int M, int N, int K, int act, float scale, int batch)
{
    dim3 g((N + BN - 1)/BN, (M + BM - 1)/BM, batch);
    gemm_wmma<<<g, 256, 0, st>>>(A, sA, lda, B, sB, ldb, bias,
                                 gate, ldg, gmap, res, ldr,
                                 oF, sF, ldoF, oH, ldoH, M, N, K, act, scale);
}

static inline void conv(hipStream_t st, const float* s, half_t* d, long long n) {
    k_convert<<<(unsigned)((n + 255)/256), 256, 0, st>>>(s, d, n);
}

extern "C" void kernel_launch(void* const* d_in, const int* in_sizes, int n_in,
                              void* d_out, int out_size, void* d_ws, size_t ws_size,
                              hipStream_t stream)
{
    const float* x_in  = (const float*)d_in[0];
    const float* noise = (const float*)d_in[1];
    const float* rawh  = (const float*)d_in[2];
    const int*   amask = (const int*)  d_in[3];
    const float* pos   = (const float*)d_in[4];
    const float* plug  = (const float*)d_in[5];
    const float* t_w0  = (const float*)d_in[6];
    const float* t_b0  = (const float*)d_in[7];
    const float* t_w2  = (const float*)d_in[8];
    const float* t_b2  = (const float*)d_in[9];
    const float* qkv_w = (const float*)d_in[10];
    const float* qkv_b = (const float*)d_in[11];
    const float* proj_w= (const float*)d_in[12];
    const float* proj_b= (const float*)d_in[13];
    const float* fc1_w = (const float*)d_in[14];
    const float* fc1_b = (const float*)d_in[15];
    const float* fc2_w = (const float*)d_in[16];
    const float* fc2_b = (const float*)d_in[17];
    const float* ada_w = (const float*)d_in[18];
    const float* ada_b = (const float*)d_in[19];
    const float* fada_w= (const float*)d_in[20];
    const float* fada_b= (const float*)d_in[21];
    const float* fin_w = (const float*)d_in[22];
    const float* fin_b = (const float*)d_in[23];
    float* out = (float*)d_out;

    // ---- workspace layout (256B aligned) ----
    size_t off = 0;
    auto alloc = [&](size_t bytes)->size_t { size_t o = off; off += (bytes + 255) & ~(size_t)255; return o; };
    char* W = (char*)d_ws;
    size_t o_tw0h  = alloc((size_t)256*CDIM*2);
    size_t o_tw2h  = alloc((size_t)CDIM*CDIM*2);
    size_t o_fadh  = alloc((size_t)CDIM*2*CDIM*2);
    size_t o_fwh   = alloc((size_t)CDIM*LATD*2);
    size_t o_wqkv  = alloc((size_t)CDIM*3*CDIM*2);
    size_t o_wproj = alloc((size_t)CDIM*CDIM*2);
    size_t o_wfc1  = alloc((size_t)CDIM*4*CDIM*2);
    size_t o_wfc2  = alloc((size_t)4*CDIM*CDIM*2);
    size_t o_wada  = alloc((size_t)CDIM*C6*2);
    size_t o_x     = alloc((size_t)NROWS*CDIM*4);
    size_t o_silu  = alloc((size_t)(MCOND+4)*CDIM*2);
    size_t o_emb   = alloc((size_t)NB*256*2);
    size_t o_th    = alloc((size_t)NB*CDIM*2);
    size_t o_mod   = alloc((size_t)(MCOND+4)*C6*4);
    size_t o_fmod  = alloc((size_t)(MCOND+4)*2*CDIM*4);
    size_t o_y     = alloc((size_t)NROWS*CDIM*2);
    size_t o_qkv   = alloc((size_t)NROWS*3*CDIM*4);
    size_t o_qp    = alloc((size_t)NB*NHEAD*NSEQ*DHP*2);
    size_t o_kt    = alloc((size_t)NB*NHEAD*DHP*NSEQ*2);
    size_t o_vp    = alloc((size_t)NB*NHEAD*NSEQ*DHP*2);
    size_t o_sc    = alloc((size_t)NHEAD*NSEQ*NSEQ*4);      // one batch's scores
    size_t o_op    = alloc((size_t)NB*NHEAD*NSEQ*DHP*4);
    size_t o_oh    = alloc((size_t)NROWS*CDIM*2);
    size_t o_h     = alloc((size_t)NROWS*4*CDIM*2);
    size_t o_xs    = alloc((size_t)NROWS*CDIM*2);
    size_t o_of    = alloc((size_t)NROWS*LATD*4);
    (void)ws_size; (void)in_sizes; (void)n_in; (void)out_size;

    half_t* tw0h = (half_t*)(W+o_tw0h);  half_t* tw2h = (half_t*)(W+o_tw2h);
    half_t* fadh = (half_t*)(W+o_fadh);  half_t* fwh  = (half_t*)(W+o_fwh);
    half_t* wqkv = (half_t*)(W+o_wqkv);  half_t* wproj= (half_t*)(W+o_wproj);
    half_t* wfc1 = (half_t*)(W+o_wfc1);  half_t* wfc2 = (half_t*)(W+o_wfc2);
    half_t* wada = (half_t*)(W+o_wada);
    float*  xbuf = (float*) (W+o_x);     half_t* siluS= (half_t*)(W+o_silu);
    half_t* embh = (half_t*)(W+o_emb);   half_t* thb  = (half_t*)(W+o_th);
    float*  modb = (float*) (W+o_mod);   float*  fmod = (float*) (W+o_fmod);
    half_t* ybuf = (half_t*)(W+o_y);     float*  qkvb = (float*) (W+o_qkv);
    half_t* qp   = (half_t*)(W+o_qp);    half_t* ktb  = (half_t*)(W+o_kt);
    half_t* vp   = (half_t*)(W+o_vp);    float*  scb  = (float*) (W+o_sc);
    float*  opb  = (float*) (W+o_op);    half_t* ohb  = (half_t*)(W+o_oh);
    half_t* hb   = (half_t*)(W+o_h);     half_t* xsb  = (half_t*)(W+o_xs);
    float*  ofull= (float*) (W+o_of);

    const int EW = (NROWS*CDIM + 255)/256;

    // persistent small weights -> f16
    conv(stream, t_w0,   tw0h, (long long)256*CDIM);
    conv(stream, t_w2,   tw2h, (long long)CDIM*CDIM);
    conv(stream, fada_w, fadh, (long long)CDIM*2*CDIM);
    conv(stream, fin_w,  fwh,  (long long)CDIM*LATD);

    // conditioning: t-embedding MLP on the 4 distinct GEN rows, silu(c) stored f16
    k_emb<<<NB, 128, 0, stream>>>(noise, embh);
    launch_gemm(stream, embh, 0, 256,  tw0h, 0, CDIM, t_b0,
                nullptr,0,0, nullptr,0, nullptr,0,0, thb, CDIM,
                NB, CDIM, 256, 1, 1.f, 1);
    launch_gemm(stream, thb, 0, CDIM,  tw2h, 0, CDIM, t_b2,
                nullptr,0,0, nullptr,0, nullptr,0,0, siluS + (size_t)HISTL*CDIM, CDIM,
                NB, CDIM, CDIM, 1, 1.f, 1);
    k_silu_plug<<<(HISTL*CDIM + 255)/256, 256, 0, stream>>>(plug, siluS);
    k_xinit<<<EW, 256, 0, stream>>>(x_in, rawh, pos, xbuf);

    const float att_scale = 1.0f / sqrtf((float)DHEAD);
    const long long lsQKV = (long long)CDIM*3*CDIM, lsP = (long long)CDIM*CDIM;
    const long long lsF1 = (long long)CDIM*4*CDIM,  lsF2 = (long long)4*CDIM*CDIM;
    const long long lsAD = (long long)CDIM*C6;

    for (int l = 0; l < 7; ++l) {
        conv(stream, qkv_w + l*lsQKV, wqkv, lsQKV);
        conv(stream, proj_w+ l*lsP,   wproj,lsP);
        conv(stream, fc1_w + l*lsF1,  wfc1, lsF1);
        conv(stream, fc2_w + l*lsF2,  wfc2, lsF2);
        conv(stream, ada_w + l*lsAD,  wada, lsAD);

        // adaLN modulation on 516 distinct rows
        launch_gemm(stream, siluS, 0, CDIM, wada, 0, C6, ada_b + (long long)l*C6,
                    nullptr,0,0, nullptr,0, modb, 0, C6, nullptr,0,
                    MCOND, C6, CDIM, 0, 1.f, 1);

        k_lnmod<<<NROWS, 256, 0, stream>>>(xbuf, modb, ybuf, 0);
        launch_gemm(stream, ybuf, 0, CDIM, wqkv, 0, 3*CDIM, qkv_b + (long long)l*3*CDIM,
                    nullptr,0,0, nullptr,0, qkvb, 0, 3*CDIM, nullptr,0,
                    NROWS, 3*CDIM, CDIM, 0, 1.f, 1);
        k_repack_qkv<<<(NB*NHEAD*NSEQ*DHP + 255)/256, 256, 0, stream>>>(qkvb, qp, ktb, vp);

        for (int b = 0; b < NB; ++b) {
            long long boff = (long long)b*NHEAD*NSEQ*DHP;
            // scores = scale * q @ k^T   (batched over 16 heads)
            launch_gemm(stream, qp + boff, (long long)NSEQ*DHP, DHP,
                        ktb + boff, (long long)DHP*NSEQ, NSEQ, nullptr,
                        nullptr,0,0, nullptr,0, scb, (long long)NSEQ*NSEQ, NSEQ, nullptr,0,
                        NSEQ, NSEQ, DHP, 0, att_scale, NHEAD);
            k_softmax<<<NHEAD*NSEQ, 256, 0, stream>>>(scb, amask);
            // o = P @ v  (P is f16 in-place inside the f32 score rows, lda = 2*NSEQ halves)
            launch_gemm(stream, (half_t*)scb, (long long)NSEQ*2*NSEQ, 2*NSEQ,
                        vp + boff, (long long)NSEQ*DHP, DHP, nullptr,
                        nullptr,0,0, nullptr,0, opb + boff, (long long)NSEQ*DHP, DHP, nullptr,0,
                        NSEQ, DHP, NSEQ, 0, 1.f, NHEAD);
        }
        k_repack_o<<<EW, 256, 0, stream>>>(opb, ohb);

        // x += g_msa * (o @ proj_w + b)
        launch_gemm(stream, ohb, 0, CDIM, wproj, 0, CDIM, proj_b + (long long)l*CDIM,
                    modb + 2*CDIM, C6, 1, xbuf, CDIM, xbuf, 0, CDIM, nullptr,0,
                    NROWS, CDIM, CDIM, 0, 1.f, 1);

        k_lnmod<<<NROWS, 256, 0, stream>>>(xbuf, modb, ybuf, 3*CDIM);
        launch_gemm(stream, ybuf, 0, CDIM, wfc1, 0, 4*CDIM, fc1_b + (long long)l*4*CDIM,
                    nullptr,0,0, nullptr,0, nullptr,0,0, hb, 4*CDIM,
                    NROWS, 4*CDIM, CDIM, 2, 1.f, 1);
        // x += g_mlp * (h @ fc2_w + b)
        launch_gemm(stream, hb, 0, 4*CDIM, wfc2, 0, CDIM, fc2_b + (long long)l*CDIM,
                    modb + 5*CDIM, C6, 1, xbuf, CDIM, xbuf, 0, CDIM, nullptr,0,
                    NROWS, CDIM, 4*CDIM, 0, 1.f, 1);
    }

    // final adaLN + head
    launch_gemm(stream, siluS, 0, CDIM, fadh, 0, 2*CDIM, fada_b,
                nullptr,0,0, nullptr,0, fmod, 0, 2*CDIM, nullptr,0,
                MCOND, 2*CDIM, CDIM, 0, 1.f, 1);
    k_finmodulate<<<EW, 256, 0, stream>>>(xbuf, fmod, xsb);
    launch_gemm(stream, xsb, 0, CDIM, fwh, 0, LATD, fin_b,
                nullptr,0,0, nullptr,0, ofull, 0, LATD, nullptr,0,
                NROWS, LATD, CDIM, 0, 1.f, 1);
    k_slice<<<(NB*GENL*LATD + 255)/256, 256, 0, stream>>>(ofull, out);
}
